// GraphConvNet_41918880809670
// MI455X (gfx1250) — matrix operand order repaired and checked by reading, not compiled, not run on hardware
//
#include <hip/hip_runtime.h>
#include <hip/hip_bf16.h>

typedef _Float16 half_t;
typedef __attribute__((ext_vector_type(16))) _Float16 v16h;
typedef __attribute__((ext_vector_type(8)))  _Float16 v8h;
typedef __attribute__((ext_vector_type(4)))  _Float16 v4h;
typedef __attribute__((ext_vector_type(2)))  _Float16 v2h;
typedef __attribute__((ext_vector_type(2)))  __fp16   v2fp16;
typedef __attribute__((ext_vector_type(8)))  float    v8f;

#define LN_EPS 1e-6f

// LDS tile strides padded +8 halves: row stride in dwords === 4 (mod 16) so the 16
// rows of a tile hit 16 disjoint bank groups (conflict-free v8h A-fragment loads).
#define STRIDE_ENC  40
#define STRIDE_H1   136
#define STRIDE_NODE 264
#define STRIDE_EDGE 392

// ---------- helpers ----------

__device__ __forceinline__ void lds_fence() {
  // LDS ops are in-order per wave; this orders the compiler + waits DScnt
  asm volatile("s_wait_dscnt 0" ::: "memory");
}

// CDNA5 hardware transcendental tanh (V_TANH_F32).
__device__ __forceinline__ float fast_tanhf(float x) {
#if __has_builtin(__builtin_amdgcn_tanhf)
  return __builtin_amdgcn_tanhf(x);
#else
  float r;
  asm volatile("v_tanh_f32 %0, %1\n\tv_nop\n\tv_nop" : "=v"(r) : "v"(x));
  return r;
#endif
}

__device__ __forceinline__ float gelu_tanh(float x) {
  const float k0 = 0.7978845608028654f; // sqrt(2/pi)
  return 0.5f * x * (1.0f + fast_tanhf(k0 * (x + 0.044715f * x * x * x)));
}

// A fragment (16x32 f16, M=lane&15): e0..7 -> K = kbase+8g+e ; e8..15 -> K = kbase+16+8g+(e-8)
__device__ __forceinline__ v16h load_a_frag(const half_t* tile, int kpad, int kbase, int lane) {
  const int g = lane >> 4, r = lane & 15;
  const half_t* p = tile + r * kpad + kbase + 8 * g;
  v8h lo = *(const v8h*)(p);
  v8h hi = *(const v8h*)(p + 16);
  return __builtin_shufflevector(lo, hi, 0,1,2,3,4,5,6,7,8,9,10,11,12,13,14,15);
}

// B fragment (32x16 f16): lane holds column n, e0..7 -> K = kbase+8g+e ; e8..15 -> K = kbase+16+8g+(e-8)
// Wt is packed transposed: Wt[n][k], row length kpad (f16)
__device__ __forceinline__ v16h load_b_frag(const half_t* Wt, int kpad, int ncolbase, int kbase, int lane) {
  const int g = lane >> 4, n = ncolbase + (lane & 15);
  const half_t* p = Wt + (size_t)n * kpad + kbase + 8 * g;
  v8h lo = *(const v8h*)(p);
  v8h hi = *(const v8h*)(p + 16);
  return __builtin_shufflevector(lo, hi, 0,1,2,3,4,5,6,7,8,9,10,11,12,13,14,15);
}

__device__ __forceinline__ v8f wmma_f16(v16h a, v16h b, v8f c) {
  return __builtin_amdgcn_wmma_f32_16x16x32_f16(false, a, false, b, (short)0, c, false, false);
}

// acc[t] += A(16 x 32*CHUNKS from LDS tile) x B(tile t of Wt); 2-deep B-fragment rotation
template<int CHUNKS>
__device__ __forceinline__ void gemm_accum(v8f (&acc)[8], const half_t* tileA, int kpadA,
                                           const half_t* Wt, int kpadW, int lane) {
  #pragma unroll
  for (int c = 0; c < CHUNKS; ++c) {
    const int kb = c * 32;
    v16h a = load_a_frag(tileA, kpadA, kb, lane);
    v16h bcur  = load_b_frag(Wt, kpadW, 0,  kb, lane);
    v16h bnext = load_b_frag(Wt, kpadW, 16, kb, lane);
    #pragma unroll
    for (int t = 0; t < 8; ++t) {
      v16h bload = bnext;
      if (t < 6) bload = load_b_frag(Wt, kpadW, (t + 2) * 16, kb, lane);
      acc[t] = wmma_f16(a, bcur, acc[t]);
      bcur = bnext;
      bnext = bload;
    }
  }
}

__device__ __forceinline__ void init_acc(v8f (&acc)[8], const float* __restrict__ b, int nc) {
  #pragma unroll
  for (int t = 0; t < 8; ++t) {
    const float bv = b[t * 16 + nc];
    #pragma unroll
    for (int i = 0; i < 8; ++i) acc[t][i] = bv;
  }
}

__device__ __forceinline__ void gelu_to_lds(const v8f (&acc)[8], half_t* th1, int lane) {
  const int g = lane >> 4, nc = lane & 15;
  #pragma unroll
  for (int t = 0; t < 8; ++t) {
    const int col = t * 16 + nc;
    #pragma unroll
    for (int v = 0; v < 8; ++v)
      th1[(v + 8 * g) * STRIDE_H1 + col] = (half_t)gelu_tanh(acc[t][v]);
  }
}

// sum across each 16-lane half of the wave32 (lane halves stay independent)
__device__ __forceinline__ float half_reduce(float x) {
  x += __shfl_xor(x, 1);
  x += __shfl_xor(x, 2);
  x += __shfl_xor(x, 4);
  x += __shfl_xor(x, 8);
  return x;
}

// pack 4 f32 -> 4 f16 (v_cvt_pk_rtz_f16_f32) and store as one 8-byte LDS write
__device__ __forceinline__ void store4h(half_t* p, float4 v) {
  v2h lo = __builtin_bit_cast(v2h, __builtin_amdgcn_cvt_pkrtz(v.x, v.y));
  v2h hi = __builtin_bit_cast(v2h, __builtin_amdgcn_cvt_pkrtz(v.z, v.w));
  *(v4h*)p = __builtin_shufflevector(lo, hi, 0, 1, 2, 3);
}

// ---------- weight packing: src [K,Nsrc] f32 row-major -> dst [Npad][Kpad] f16 transposed ----------

__global__ void pack_wt(const float* __restrict__ src, half_t* __restrict__ dst,
                        int Kuse, int Kpad, int Nsrc, int Npad) {
  int i = blockIdx.x * blockDim.x + threadIdx.x;
  if (i >= Npad * Kpad) return;
  int n = i / Kpad, k = i % Kpad;
  float v = (k < Kuse && n < Nsrc) ? src[(size_t)k * Nsrc + n] : 0.0f;
  dst[i] = (half_t)v;
}

// fold globals (2 values, single graph) into layer-1 bias: rows Kuse, Kuse+1 of W0
__global__ void fold_bias(const float* __restrict__ b0, const float* __restrict__ W0,
                          int Kuse, const float* __restrict__ g, float* __restrict__ dst) {
  int h = threadIdx.x; // 128 threads
  dst[h] = b0[h] + g[0] * W0[(size_t)Kuse * 128 + h] + g[1] * W0[(size_t)(Kuse + 1) * 128 + h];
}

// ---------- encoder: in[rows,F] -> gelu(in@W0+b0)@W1+b1 -> out[rows,128] ----------

template<int F>
__global__ void __launch_bounds__(128) encode_kernel(
    const float* __restrict__ in, const float* __restrict__ b0, const half_t* __restrict__ W0t,
    const float* __restrict__ b1, const half_t* __restrict__ W1t,
    float* __restrict__ out, int nrows) {
  __shared__ __attribute__((aligned(64))) half_t s_in[4][16 * STRIDE_ENC];
  __shared__ __attribute__((aligned(64))) half_t s_h1[4][16 * STRIDE_H1];
  const int wave = threadIdx.x >> 5, lane = threadIdx.x & 31;
  const int tile = blockIdx.x * 4 + wave;
  if (tile >= (nrows >> 4)) return;
  const int rbase = tile << 4;
  half_t* tin = s_in[wave];
  half_t* th1 = s_h1[wave];
  if (lane < 16) {
    const float* src = in + (size_t)(rbase + lane) * F;
    #pragma unroll
    for (int j = 0; j < 32; ++j)
      tin[lane * STRIDE_ENC + j] = (half_t)(j < F ? src[j] : 0.0f);
  }
  lds_fence();
  const int g = lane >> 4, nc = lane & 15;
  // layer 1: K = 32 (zero padded)
  v8f acc[8];
  init_acc(acc, b0, nc);
  gemm_accum<1>(acc, tin, STRIDE_ENC, W0t, 32, lane);
  gelu_to_lds(acc, th1, lane);
  lds_fence();
  // layer 2: K = 128
  init_acc(acc, b1, nc);
  gemm_accum<4>(acc, th1, STRIDE_H1, W1t, 128, lane);
  #pragma unroll
  for (int t = 0; t < 8; ++t) {
    const int col = t * 16 + nc;
    #pragma unroll
    for (int v = 0; v < 8; ++v)
      out[(size_t)(rbase + v + 8 * g) * 128 + col] = acc[t][v];
  }
}

// ---------- edge update (fused MLP + segment-sum atomics + skip + LayerNorm) ----------

__global__ void __launch_bounds__(64) edge_update_kernel(
    const float* __restrict__ nh, float* __restrict__ eh,
    const int* __restrict__ snd, const int* __restrict__ rcv,
    const float* __restrict__ b0, const half_t* __restrict__ W0t,   // [128][384]
    const float* __restrict__ b1, const half_t* __restrict__ W1t,   // [128][128]
    const float* __restrict__ ln_s, const float* __restrict__ ln_b,
    float* __restrict__ received, int nedges) {
  __shared__ __attribute__((aligned(64))) half_t s_in[2][16 * STRIDE_EDGE];
  __shared__ __attribute__((aligned(64))) half_t s_h1[2][16 * STRIDE_H1];
  const int wave = threadIdx.x >> 5, lane = threadIdx.x & 31;
  const int tile = blockIdx.x * 2 + wave;
  if (tile >= (nedges >> 4)) return;
  const int ebase = tile << 4;
  half_t* tin = s_in[wave];
  half_t* th1 = s_h1[wave];
  // stage 16 x [eh | nh[snd] | nh[rcv]] f16 (nh gathers hit L2: nh is 25.6MB)
  // snd/rcv indices are wave-uniform here -> scalar (SMEM) loads
  for (int r = 0; r < 16; ++r) {
    const int si = snd[ebase + r], ri = rcv[ebase + r];
    float4 e4 = *(const float4*)(eh + (size_t)(ebase + r) * 128 + lane * 4);
    float4 s4 = *(const float4*)(nh + (size_t)si * 128 + lane * 4);
    float4 r4 = *(const float4*)(nh + (size_t)ri * 128 + lane * 4);
    half_t* row = tin + r * STRIDE_EDGE;
    store4h(row + lane * 4, e4);
    store4h(row + 128 + lane * 4, s4);
    store4h(row + 256 + lane * 4, r4);
  }
  lds_fence();
  const int g = lane >> 4, nc = lane & 15;
  // layer 1: K = 384 (globals folded into b0)
  v8f acc[8];
  init_acc(acc, b0, nc);
  gemm_accum<12>(acc, tin, STRIDE_EDGE, W0t, 384, lane);
  gelu_to_lds(acc, th1, lane);
  lds_fence();
  // layer 2: K = 128
  init_acc(acc, b1, nc);
  gemm_accum<4>(acc, th1, STRIDE_H1, W1t, 128, lane);
  // epilogue: atomic segment-sum of new_e, then eh = LN(eh + new_e)
  int rr[8];
  #pragma unroll
  for (int v = 0; v < 8; ++v) rr[v] = rcv[ebase + v + 8 * g];
  float s[8], q[8];
  #pragma unroll
  for (int v = 0; v < 8; ++v) { s[v] = 0.f; q[v] = 0.f; }
  #pragma unroll
  for (int t = 0; t < 8; ++t) {
    const int col = t * 16 + nc;
    #pragma unroll
    for (int v = 0; v < 8; ++v) {
      float ne = acc[t][v];
      atomicAdd(received + (size_t)rr[v] * 128 + col, ne);
      float y = ne + eh[(size_t)(ebase + v + 8 * g) * 128 + col];
      acc[t][v] = y;
      s[v] += y; q[v] += y * y;
    }
  }
  #pragma unroll
  for (int v = 0; v < 8; ++v) {
    float m = half_reduce(s[v]) * (1.0f / 128.0f);
    float var = half_reduce(q[v]) * (1.0f / 128.0f) - m * m;
    s[v] = m;
    q[v] = rsqrtf(var + LN_EPS);
  }
  #pragma unroll
  for (int t = 0; t < 8; ++t) {
    const int col = t * 16 + nc;
    const float sc = ln_s[col], bi = ln_b[col];
    #pragma unroll
    for (int v = 0; v < 8; ++v)
      eh[(size_t)(ebase + v + 8 * g) * 128 + col] = (acc[t][v] - s[v]) * q[v] * sc + bi;
  }
}

// ---------- node update (fused MLP + skip + LayerNorm) ----------

__global__ void __launch_bounds__(128) node_update_kernel(
    float* __restrict__ nh, const float* __restrict__ received,
    const float* __restrict__ b0, const half_t* __restrict__ W0t,   // [128][256]
    const float* __restrict__ b1, const half_t* __restrict__ W1t,   // [128][128]
    const float* __restrict__ ln_s, const float* __restrict__ ln_b, int nrows) {
  __shared__ __attribute__((aligned(64))) half_t s_in[4][16 * STRIDE_NODE];
  __shared__ __attribute__((aligned(64))) half_t s_h1[4][16 * STRIDE_H1];
  const int wave = threadIdx.x >> 5, lane = threadIdx.x & 31;
  const int tile = blockIdx.x * 4 + wave;
  if (tile >= (nrows >> 4)) return;
  const int rbase = tile << 4;
  half_t* tin = s_in[wave];
  half_t* th1 = s_h1[wave];
  for (int r = 0; r < 16; ++r) {
    float4 x = *(const float4*)(nh + (size_t)(rbase + r) * 128 + lane * 4);
    float4 y = *(const float4*)(received + (size_t)(rbase + r) * 128 + lane * 4);
    half_t* row = tin + r * STRIDE_NODE;
    store4h(row + lane * 4, x);
    store4h(row + 128 + lane * 4, y);
  }
  lds_fence();
  const int g = lane >> 4, nc = lane & 15;
  v8f acc[8];
  init_acc(acc, b0, nc);
  gemm_accum<8>(acc, tin, STRIDE_NODE, W0t, 256, lane);
  gelu_to_lds(acc, th1, lane);
  lds_fence();
  init_acc(acc, b1, nc);
  gemm_accum<4>(acc, th1, STRIDE_H1, W1t, 128, lane);
  float s[8], q[8];
  #pragma unroll
  for (int v = 0; v < 8; ++v) { s[v] = 0.f; q[v] = 0.f; }
  #pragma unroll
  for (int t = 0; t < 8; ++t) {
    const int col = t * 16 + nc;
    #pragma unroll
    for (int v = 0; v < 8; ++v) {
      float y = acc[t][v] + nh[(size_t)(rbase + v + 8 * g) * 128 + col];
      acc[t][v] = y;
      s[v] += y; q[v] += y * y;
    }
  }
  #pragma unroll
  for (int v = 0; v < 8; ++v) {
    float m = half_reduce(s[v]) * (1.0f / 128.0f);
    float var = half_reduce(q[v]) * (1.0f / 128.0f) - m * m;
    s[v] = m;
    q[v] = rsqrtf(var + LN_EPS);
  }
  #pragma unroll
  for (int t = 0; t < 8; ++t) {
    const int col = t * 16 + nc;
    const float sc = ln_s[col], bi = ln_b[col];
    #pragma unroll
    for (int v = 0; v < 8; ++v)
      nh[(size_t)(rbase + v + 8 * g) * 128 + col] = (acc[t][v] - s[v]) * q[v] * sc + bi;
  }
}

// ---------- decoder: out[rows,3] = gelu(nh@W0+b0)@W1+b1 ----------

__global__ void __launch_bounds__(128) decode_kernel(
    const float* __restrict__ nh,
    const float* __restrict__ b0, const half_t* __restrict__ W0t,   // [128][128]
    const float* __restrict__ b1, const half_t* __restrict__ W1t,   // [16][128] (3 valid cols)
    float* __restrict__ out, int nrows) {
  __shared__ __attribute__((aligned(64))) half_t s_in[4][16 * STRIDE_H1];
  __shared__ __attribute__((aligned(64))) half_t s_h1[4][16 * STRIDE_H1];
  const int wave = threadIdx.x >> 5, lane = threadIdx.x & 31;
  const int tile = blockIdx.x * 4 + wave;
  if (tile >= (nrows >> 4)) return;
  const int rbase = tile << 4;
  half_t* tin = s_in[wave];
  half_t* th1 = s_h1[wave];
  for (int r = 0; r < 16; ++r) {
    float4 x = *(const float4*)(nh + (size_t)(rbase + r) * 128 + lane * 4);
    store4h(tin + r * STRIDE_H1 + lane * 4, x);
  }
  lds_fence();
  const int g = lane >> 4, nc = lane & 15;
  v8f acc[8];
  init_acc(acc, b0, nc);
  gemm_accum<4>(acc, tin, STRIDE_H1, W0t, 128, lane);
  gelu_to_lds(acc, th1, lane);
  lds_fence();
  // layer 2: one 16-col tile, columns 0..2 valid
  v8f c2;
  const float bv = (nc < 3) ? b1[nc] : 0.0f;
  #pragma unroll
  for (int i = 0; i < 8; ++i) c2[i] = bv;
  #pragma unroll
  for (int c = 0; c < 4; ++c)
    c2 = wmma_f16(load_a_frag(th1, STRIDE_H1, c * 32, lane),
                  load_b_frag(W1t, 128, 0, c * 32, lane), c2);
  if (nc < 3) {
    #pragma unroll
    for (int v = 0; v < 8; ++v)
      out[(size_t)(rbase + v + 8 * g) * 3 + nc] = c2[v];
  }
}

// ---------- host ----------

extern "C" void kernel_launch(void* const* d_in, const int* in_sizes, int n_in,
                              void* d_out, int out_size, void* d_ws, size_t ws_size,
                              hipStream_t stream) {
  (void)n_in; (void)out_size; (void)ws_size;
  const float* nodes    = (const float*)d_in[0];
  const float* edges    = (const float*)d_in[1];
  const float* globals_ = (const float*)d_in[2];
  const int*   senders  = (const int*)d_in[3];
  const int*   receivers= (const int*)d_in[4];
  const float* ne_W0 = (const float*)d_in[5];  const float* ne_b0 = (const float*)d_in[6];
  const float* ne_W1 = (const float*)d_in[7];  const float* ne_b1 = (const float*)d_in[8];
  const float* ee_W0 = (const float*)d_in[9];  const float* ee_b0 = (const float*)d_in[10];
  const float* ee_W1 = (const float*)d_in[11]; const float* ee_b1 = (const float*)d_in[12];
  const float* eu_W0 = (const float*)d_in[13]; const float* eu_b0 = (const float*)d_in[14];
  const float* eu_W1 = (const float*)d_in[15]; const float* eu_b1 = (const float*)d_in[16];
  const float* nu_W0 = (const float*)d_in[17]; const float* nu_b0 = (const float*)d_in[18];
  const float* nu_W1 = (const float*)d_in[19]; const float* nu_b1 = (const float*)d_in[20];
  const float* ln_s  = (const float*)d_in[21]; const float* ln_b  = (const float*)d_in[22];
  const float* dec_W0= (const float*)d_in[23]; const float* dec_b0= (const float*)d_in[24];
  const float* dec_W1= (const float*)d_in[25]; const float* dec_b1= (const float*)d_in[26];

  const int N_ = in_sizes[0] / 7;   // NODE_F = 7
  const int E_ = in_sizes[1] / 3;   // EDGE_F = 3

  char* wp = (char*)d_ws;
  auto carve = [&](size_t bytes) -> void* {
    void* r = (void*)wp;
    wp += (bytes + 255) & ~(size_t)255;
    return r;
  };
  float*  nh   = (float*)carve((size_t)N_ * 128 * 4);
  float*  eh   = (float*)carve((size_t)E_ * 128 * 4);
  float*  recv = (float*)carve((size_t)N_ * 128 * 4);
  half_t* neW0t = (half_t*)carve(128 * 32 * 2);
  half_t* neW1t = (half_t*)carve(128 * 128 * 2);
  half_t* eeW0t = (half_t*)carve(128 * 32 * 2);
  half_t* eeW1t = (half_t*)carve(128 * 128 * 2);
  half_t* euW0t[2]; half_t* euW1t[2]; half_t* nuW0t[2]; half_t* nuW1t[2];
  float*  eub0e[2]; float* nub0e[2];
  for (int s = 0; s < 2; ++s) {
    euW0t[s] = (half_t*)carve(128 * 384 * 2);
    euW1t[s] = (half_t*)carve(128 * 128 * 2);
    nuW0t[s] = (half_t*)carve(128 * 256 * 2);
    nuW1t[s] = (half_t*)carve(128 * 128 * 2);
    eub0e[s] = (float*)carve(128 * 4);
    nub0e[s] = (float*)carve(128 * 4);
  }
  half_t* decW0t = (half_t*)carve(128 * 128 * 2);
  half_t* decW1t = (half_t*)carve(16 * 128 * 2);

  auto pack = [&](const float* src, half_t* dst, int Kuse, int Kpad, int Nsrc, int Npad) {
    int n = Npad * Kpad;
    pack_wt<<<(n + 255) / 256, 256, 0, stream>>>(src, dst, Kuse, Kpad, Nsrc, Npad);
  };
  pack(ne_W0, neW0t, 7, 32, 128, 128);
  pack(ne_W1, neW1t, 128, 128, 128, 128);
  pack(ee_W0, eeW0t, 3, 32, 128, 128);
  pack(ee_W1, eeW1t, 128, 128, 128, 128);
  for (int s = 0; s < 2; ++s) {
    pack(eu_W0 + (size_t)s * 386 * 128, euW0t[s], 384, 384, 128, 128);
    pack(eu_W1 + (size_t)s * 128 * 128, euW1t[s], 128, 128, 128, 128);
    pack(nu_W0 + (size_t)s * 258 * 128, nuW0t[s], 256, 256, 128, 128);
    pack(nu_W1 + (size_t)s * 128 * 128, nuW1t[s], 128, 128, 128, 128);
    fold_bias<<<1, 128, 0, stream>>>(eu_b0 + s * 128, eu_W0 + (size_t)s * 386 * 128, 384, globals_, eub0e[s]);
    fold_bias<<<1, 128, 0, stream>>>(nu_b0 + s * 128, nu_W0 + (size_t)s * 258 * 128, 256, globals_, nub0e[s]);
  }
  pack(dec_W0, decW0t, 128, 128, 128, 128);
  pack(dec_W1, decW1t, 128, 128, 3, 16);

  const int n_tiles = N_ >> 4;  // 3125
  const int e_tiles = E_ >> 4;  // 31250
  const int n_blocks = (n_tiles + 3) / 4;
  const int e_blocks4 = (e_tiles + 3) / 4;
  const int e_blocks2 = (e_tiles + 1) / 2;

  encode_kernel<7><<<n_blocks, 128, 0, stream>>>(nodes, ne_b0, neW0t, ne_b1, neW1t, nh, N_);
  encode_kernel<3><<<e_blocks4, 128, 0, stream>>>(edges, ee_b0, eeW0t, ee_b1, eeW1t, eh, E_);

  for (int s = 0; s < 2; ++s) {
    (void)hipMemsetAsync(recv, 0, (size_t)N_ * 128 * 4, stream);
    edge_update_kernel<<<e_blocks2, 64, 0, stream>>>(
        nh, eh, senders, receivers, eub0e[s], euW0t[s], eu_b1 + s * 128, euW1t[s],
        ln_s, ln_b, recv, E_);
    node_update_kernel<<<n_blocks, 128, 0, stream>>>(
        nh, recv, nub0e[s], nuW0t[s], nu_b1 + s * 128, nuW1t[s], ln_s, ln_b, N_);
  }

  decode_kernel<<<n_blocks, 128, 0, stream>>>(nh, dec_b0, decW0t, dec_b1, decW1t, (float*)d_out, N_);
}
